// GIN_23055384445759
// MI455X (gfx1250) — compile-verified, matrix operand order checked by reading, not run on hardware
//
#include <hip/hip_runtime.h>
#include <hip/hip_bf16.h>

typedef __attribute__((ext_vector_type(16))) __bf16 v16bf;
typedef __attribute__((ext_vector_type(8)))  float  v8f;

#define DEVFN __device__ __forceinline__

// ---- helpers -------------------------------------------------------------

// round-to-nearest-even f32 -> bf16, packed pair into one dword
DEVFN unsigned int f2bf_pk(float lo, float hi) {
  union { float f; unsigned int u; } a, b;
  a.f = lo; b.f = hi;
  unsigned int rl = (a.u + 0x7FFFu + ((a.u >> 16) & 1u)) >> 16;
  unsigned int rh = (b.u + 0x7FFFu + ((b.u >> 16) & 1u)) >> 16;
  return (rl & 0xFFFFu) | (rh << 16);
}

DEVFN void atomAddF(float* p, float v) {
  (void)__hip_atomic_fetch_add(p, v, __ATOMIC_RELAXED, __HIP_MEMORY_SCOPE_AGENT);
}

// Load one 16x32 bf16 WMMA operand fragment from an LDS row (ISA layout:
// lanes 0-15 hold k 0..7 & 16..23, lanes 16-31 hold k 8..15 & 24..31).
DEVFN v16bf load_frag(const unsigned int* rowPtr, int kb, int kHi) {
  union { unsigned int u[8]; v16bf v; } r;
  const int b0 = (kb + kHi) >> 1;
  const int b1 = (kb + kHi + 16) >> 1;
#pragma unroll
  for (int j = 0; j < 4; ++j) {
    r.u[j]     = rowPtr[b0 + j];
    r.u[4 + j] = rowPtr[b1 + j];
  }
  return r.v;
}

// ---- elementwise: out = (1+eps)*in --------------------------------------

__global__ __launch_bounds__(256)
void gin_scale_copy(const float* __restrict__ in, float* __restrict__ out,
                    const float* __restrict__ eps, int n4) {
  int i = blockIdx.x * 256 + threadIdx.x;
  if (i >= n4) return;
  float s = 1.0f + eps[0];
  float4 v = ((const float4*)in)[i];
  v.x *= s; v.y *= s; v.z *= s; v.w *= s;
  ((float4*)out)[i] = v;
}

__global__ __launch_bounds__(256)
void gin_zero(float* __restrict__ p, int n) {
  int i = blockIdx.x * 256 + threadIdx.x;
  if (i < n) p[i] = 0.0f;
}

// ---- edge scatter: out[dst] += x[src], one wave per edge ----------------

template<int FDIM>
__global__ __launch_bounds__(256)
void gin_scatter(const float* __restrict__ x, const int* __restrict__ ei,
                 float* __restrict__ out, int nE) {
  int e = blockIdx.x * 8 + (threadIdx.x >> 5);
  if (e >= nE) return;
  int lane = threadIdx.x & 31;
  int s = ei[e];
  int d = ei[nE + e];
  const float* xs = x + (size_t)s * FDIM + lane * 4;
  float* od = out + (size_t)d * FDIM + lane * 4;
#pragma unroll
  for (int c = 0; c < FDIM / 128; ++c) {
    float4 v = *(const float4*)(xs + c * 128);
    atomAddF(od + c * 128 + 0, v.x);
    atomAddF(od + c * 128 + 1, v.y);
    atomAddF(od + c * 128 + 2, v.z);
    atomAddF(od + c * 128 + 3, v.w);
  }
}

// ---- BN stats -> per-channel affine (a = g*rsqrt(var+eps), b = be - mu*a)

__global__ __launch_bounds__(256)
void gin_bn_affine(const float* __restrict__ sum, const float* __restrict__ sq,
                   const float* __restrict__ gamma, const float* __restrict__ beta,
                   float* __restrict__ aOut, float* __restrict__ bOut, float invN) {
  int c = threadIdx.x;
  float mu  = sum[c] * invN;
  float var = sq[c] * invN - mu * mu;
  float inv = rsqrtf(var + 1e-5f);
  float a = gamma[c] * inv;
  aOut[c] = a;
  bOut[c] = beta[c] - mu * a;
}

// ---- fused WMMA GEMM: Y = act( (affine?(a*A+b):A) @ W + bias ) ----------
// 64x64 output tile, 8 waves, each wave: one 16-row A frag x two 16-col B
// frags per 32-wide k-step. K staged in bf16 LDS chunks of 64.
// Optional epilogues: per-column sum/sumsq atomics (BN stats), log_softmax.

template<int K, bool AFFINE, bool RELU, bool STATS, bool LSM>
__global__ __launch_bounds__(256)
void gin_gemm(const float* __restrict__ A, const float* __restrict__ W,
              const float* __restrict__ bias,
              const float* __restrict__ affA, const float* __restrict__ affB,
              float* __restrict__ Y,
              float* __restrict__ sumOut, float* __restrict__ sqOut,
              int Nrows, int NOUT) {
  constexpr int KW = 66;                    // 64 k + 2 pad (bf16 elems)
  constexpr int TW = 68;                    // f32 tile row stride
  constexpr int AB_BYTES   = 2 * 64 * KW * 2;
  constexpr int TILE_BYTES = (64 * TW + 512) * 4;
  constexpr int SMEM_BYTES = AB_BYTES > TILE_BYTES ? AB_BYTES : TILE_BYTES;
  __shared__ __align__(16) char smem[SMEM_BYTES];
  unsigned short* lA = (unsigned short*)smem;
  unsigned short* lB = lA + 64 * KW;

  const int tid  = threadIdx.x;
  const int lane = tid & 31;
  const int wave = tid >> 5;
  const int wm = wave & 3;                  // 4 wave rows of 16
  const int wn = wave >> 2;                 // 2 wave cols of 32
  const int rowBase = blockIdx.x * 64;
  const int colBase = blockIdx.y * 64;

  const int kHi = (lane >> 4) * 8;
  const unsigned int* aRow  = (const unsigned int*)(lA + (wm * 16 + (lane & 15)) * KW);
  const unsigned int* bRow0 = (const unsigned int*)(lB + (wn * 32 +      (lane & 15)) * KW);
  const unsigned int* bRow1 = (const unsigned int*)(lB + (wn * 32 + 16 + (lane & 15)) * KW);

  v8f acc0 = {};
  v8f acc1 = {};

  for (int k0 = 0; k0 < K; k0 += 64) {
    __syncthreads();
    // stage A chunk [64 rows][64 k] as bf16 (optionally BN-affine applied)
#pragma unroll
    for (int it = 0; it < 4; ++it) {
      int i   = tid + it * 256;             // 0..1023
      int row = i >> 4;
      int c4  = (i & 15) * 4;
      int gr  = rowBase + row;
      int gk  = k0 + c4;
      float4 v = make_float4(0.f, 0.f, 0.f, 0.f);
      if (gr < Nrows) v = *(const float4*)(A + (size_t)gr * K + gk);
      if (AFFINE) {
        v.x = fmaf(v.x, affA[gk + 0], affB[gk + 0]);
        v.y = fmaf(v.y, affA[gk + 1], affB[gk + 1]);
        v.z = fmaf(v.z, affA[gk + 2], affB[gk + 2]);
        v.w = fmaf(v.w, affA[gk + 3], affB[gk + 3]);
      }
      unsigned int* dp = (unsigned int*)(lA + row * KW + c4);
      dp[0] = f2bf_pk(v.x, v.y);
      dp[1] = f2bf_pk(v.z, v.w);
    }
    // stage B chunk transposed: lB[n][k], W is [K,NOUT] row-major
#pragma unroll
    for (int it = 0; it < 4; ++it) {
      int i  = tid + it * 256;
      int n  = i >> 4;
      int c4 = (i & 15) * 4;
      int gc = colBase + n;
      int gk = k0 + c4;
      float w0 = W[(size_t)(gk + 0) * NOUT + gc];
      float w1 = W[(size_t)(gk + 1) * NOUT + gc];
      float w2 = W[(size_t)(gk + 2) * NOUT + gc];
      float w3 = W[(size_t)(gk + 3) * NOUT + gc];
      unsigned int* dp = (unsigned int*)(lB + n * KW + c4);
      dp[0] = f2bf_pk(w0, w1);
      dp[1] = f2bf_pk(w2, w3);
    }
    __syncthreads();
#pragma unroll
    for (int kb = 0; kb < 64; kb += 32) {
      v16bf af = load_frag(aRow,  kb, kHi);
      v16bf b0 = load_frag(bRow0, kb, kHi);
      v16bf b1 = load_frag(bRow1, kb, kHi);
      acc0 = __builtin_amdgcn_wmma_f32_16x16x32_bf16(false, af, false, b0,
                                                     (short)0, acc0, false, false);
      acc1 = __builtin_amdgcn_wmma_f32_16x16x32_bf16(false, af, false, b1,
                                                     (short)0, acc1, false, false);
    }
  }
  __syncthreads();

  // spill accumulators (+bias, +relu) into f32 LDS tile for epilogues
  float* tile = (float*)smem;
  {
    const int rOff = (lane >> 4) * 8;
    const int cL   = lane & 15;
#pragma unroll
    for (int r = 0; r < 8; ++r) {
      int row = wm * 16 + rOff + r;
      int c0  = wn * 32 + cL;
      float v0 = acc0[r] + bias[colBase + c0];
      float v1 = acc1[r] + bias[colBase + c0 + 16];
      if (RELU) { v0 = fmaxf(v0, 0.f); v1 = fmaxf(v1, 0.f); }
      tile[row * TW + c0]      = v0;
      tile[row * TW + c0 + 16] = v1;
    }
  }
  __syncthreads();

  if (!LSM) {
#pragma unroll
    for (int it = 0; it < 16; ++it) {
      int i   = tid + it * 256;             // 4096 elems
      int row = i >> 6;
      int col = i & 63;
      int gr  = rowBase + row;
      if (gr < Nrows) Y[(size_t)gr * NOUT + colBase + col] = tile[row * TW + col];
    }
  }

  if (STATS) {
    float* part = tile + 64 * TW;           // 512 floats scratch
    int c = tid & 63;
    int q = tid >> 6;
    float s = 0.f, s2 = 0.f;
    for (int r = q * 16; r < q * 16 + 16; ++r) {
      if (rowBase + r < Nrows) {
        float v = tile[r * TW + c];
        s += v; s2 += v * v;
      }
    }
    part[tid]       = s;
    part[256 + tid] = s2;
    __syncthreads();
    if (q == 0) {
      float ts = 0.f, ts2 = 0.f;
#pragma unroll
      for (int j = 0; j < 4; ++j) {
        ts  += part[j * 64 + c];
        ts2 += part[256 + j * 64 + c];
      }
      atomAddF(&sumOut[colBase + c], ts);
      atomAddF(&sqOut[colBase + c], ts2);
    }
  }

  if (LSM) {                                // NOUT==64, whole row in tile
    if (tid < 64) {
      int gr = rowBase + tid;
      if (gr < Nrows) {
        const float* rp = tile + tid * TW;
        float m = rp[0];
#pragma unroll
        for (int j = 1; j < 64; ++j) m = fmaxf(m, rp[j]);
        float s = 0.f;
#pragma unroll
        for (int j = 0; j < 64; ++j) s += __expf(rp[j] - m);
        float lse = m + __logf(s);
#pragma unroll
        for (int j = 0; j < 64; ++j) Y[(size_t)gr * 64 + j] = rp[j] - lse;
      }
    }
  }
}

// ---- driver --------------------------------------------------------------

extern "C" void kernel_launch(void* const* d_in, const int* in_sizes, int n_in,
                              void* d_out, int out_size, void* d_ws, size_t ws_size,
                              hipStream_t stream) {
  const float* x    = (const float*)d_in[0];
  const int*   ei   = (const int*)d_in[1];
  const float* eps1 = (const float*)d_in[2];
  const float* W1a  = (const float*)d_in[3];
  const float* b1a  = (const float*)d_in[4];
  const float* g1   = (const float*)d_in[5];
  const float* be1  = (const float*)d_in[6];
  const float* W1b  = (const float*)d_in[7];
  const float* b1b  = (const float*)d_in[8];
  const float* eps2 = (const float*)d_in[9];
  const float* W2a  = (const float*)d_in[10];
  const float* b2a  = (const float*)d_in[11];
  const float* g2   = (const float*)d_in[12];
  const float* be2  = (const float*)d_in[13];
  const float* W2b  = (const float*)d_in[14];
  const float* b2b  = (const float*)d_in[15];

  const int N = in_sizes[0] / 128;
  const int E = in_sizes[1] / 2;

  size_t nb = (size_t)N * 256;
  float* bufA  = (float*)d_ws;              // N x 256 (first pass uses 128 cols)
  float* bufB  = bufA + nb;                 // N x 256
  float* stats = bufB + nb;                 // 2048 floats
  float* sum1 = stats;        float* sq1 = stats + 256;
  float* a1   = stats + 512;  float* b1f = stats + 768;
  float* sum2 = stats + 1024; float* sq2 = stats + 1280;
  float* a2   = stats + 1536; float* b2f = stats + 1792;

  const float invN = 1.0f / (float)N;
  const int gemmBlocksM = (N + 63) / 64;
  const int scatterBlocks = (E + 7) / 8;

  // zero BN accumulators
  gin_zero<<<8, 256, 0, stream>>>(stats, 2048);

  // ---- layer 1 ----
  // h_in = (1+eps1)*x ; h_in += scatter(x)
  gin_scale_copy<<<(N * 128 / 4 + 255) / 256, 256, 0, stream>>>(x, bufA, eps1, N * 128 / 4);
  gin_scatter<128><<<scatterBlocks, 256, 0, stream>>>(x, ei, bufA, E);
  // h1 = relu(h_in @ W1a + b1a), accumulate BN stats
  gin_gemm<128, false, true, true, false>
      <<<dim3(gemmBlocksM, 4), 256, 0, stream>>>(bufA, W1a, b1a, nullptr, nullptr,
                                                 bufB, sum1, sq1, N, 256);
  gin_bn_affine<<<1, 256, 0, stream>>>(sum1, sq1, g1, be1, a1, b1f, invN);
  // hrelu = relu(BN(h1) @ W1b + b1b)   (conv1 output + outer relu)
  gin_gemm<256, true, true, false, false>
      <<<dim3(gemmBlocksM, 4), 256, 0, stream>>>(bufB, W1b, b1b, a1, b1f,
                                                 bufA, nullptr, nullptr, N, 256);

  // ---- layer 2 ----
  gin_scale_copy<<<(N * 256 / 4 + 255) / 256, 256, 0, stream>>>(bufA, bufB, eps2, N * 256 / 4);
  gin_scatter<256><<<scatterBlocks, 256, 0, stream>>>(bufA, ei, bufB, E);
  gin_gemm<256, false, true, true, false>
      <<<dim3(gemmBlocksM, 4), 256, 0, stream>>>(bufB, W2a, b2a, nullptr, nullptr,
                                                 bufA, sum2, sq2, N, 256);
  gin_bn_affine<<<1, 256, 0, stream>>>(sum2, sq2, g2, be2, a2, b2f, invN);
  // out = log_softmax(BN(t) @ W2b + b2b)
  gin_gemm<256, true, false, false, true>
      <<<dim3(gemmBlocksM, 1), 256, 0, stream>>>(bufA, W2b, b2b, a2, b2f,
                                                 (float*)d_out, nullptr, nullptr, N, 64);
  (void)n_in; (void)out_size; (void)ws_size;
}
